// RNNType_59141699666019
// MI455X (gfx1250) — matrix operand order repaired and checked by reading, not compiled, not run on hardware
//
#include <hip/hip_runtime.h>
#include <hip/hip_bf16.h>

#define VOC 32000
#define EMB 128
#define HID 256
#define BAT 16
#define SEQ 256
#define G4H 1024  // 4*HID

typedef __bf16 bf16t;
typedef __attribute__((ext_vector_type(16))) __bf16 v16bf;
typedef __attribute__((ext_vector_type(8)))  __bf16 v8bf;
typedef __attribute__((ext_vector_type(8)))  float  v8f;

static __device__ __forceinline__ v8f wmma_bf16(v16bf a, v16bf b, v8f c) {
  // (neg_a, A, neg_b, B, c_mod, C, reuse_a, reuse_b)
  return __builtin_amdgcn_wmma_f32_16x16x32_bf16(false, a, false, b, (short)0, c,
                                                 false, false);
}

// ---- 16-bit A fragment (16x32), source row-major with leading dim ld ----
// lane<16: row m=lane,   K = {k0..k0+7} U {k0+16..k0+23}
// lane>=16: row m=lane-16, K = {k0+8..k0+15} U {k0+24..k0+31}
static __device__ __forceinline__ v16bf fragA(const bf16t* mat, int row, int ld,
                                              int k0, int lane) {
  int kb = (lane & 16) ? 8 : 0;
  const bf16t* p = mat + (long)row * ld + k0 + kb;
  v8bf lo = *(const v8bf*)(p);
  v8bf hi = *(const v8bf*)(p + 16);
  v16bf f;
#pragma unroll
  for (int i = 0; i < 8; ++i) { f[i] = lo[i]; f[i + 8] = hi[i]; }
  return f;
}

// A fragment from an f32 row (convert to bf16 in-register)
static __device__ __forceinline__ v16bf fragA_f32(const float* rowp, int k0, int lane) {
  int kb = (lane & 16) ? 8 : 0;
  const float* p = rowp + k0 + kb;
  v8f lo = *(const v8f*)(p);
  v8f hi = *(const v8f*)(p + 16);
  v16bf f;
#pragma unroll
  for (int i = 0; i < 8; ++i) { f[i] = (bf16t)lo[i]; f[i + 8] = (bf16t)hi[i]; }
  return f;
}

// ---- B fragment (32x16) from W stored [N][K] row-major (i.e. B^T) ----
// lane<16: col n=lane,   K = {k0..k0+7} U {k0+16..k0+23} ; lane>=16: col n=lane-16, K = {k0+8..k0+15} U {k0+24..k0+31}
static __device__ __forceinline__ v16bf fragB(const bf16t* w, int nrow, int ld,
                                              int k0, int lane) {
  int kb = (lane & 16) ? 8 : 0;
  const bf16t* p = w + (long)nrow * ld + k0 + kb;
  v8bf lo = *(const v8bf*)(p);
  v8bf hi = *(const v8bf*)(p + 16);
  v16bf f;
#pragma unroll
  for (int i = 0; i < 8; ++i) { f[i] = lo[i]; f[i + 8] = hi[i]; }
  return f;
}

static __device__ __forceinline__ float sigm(float x) {
  return 1.f / (1.f + __expf(-x));
}
static __device__ __forceinline__ float fast_tanh(float x) {
  float e = __expf(2.f * x);
  return (e - 1.f) / (e + 1.f);
}

// -------------------- weight prep: f32 -> bf16, fold biases --------------------
__global__ __launch_bounds__(256) void k_prep(
    const float* Wih0, const float* Whh0, const float* Wih1, const float* Whh1,
    const float* Wfc, const float* bih0, const float* bhh0, const float* bih1,
    const float* bhh1, bf16t* wih0, bf16t* whh0, bf16t* wih1, bf16t* whh1,
    bf16t* wfc, float* b0, float* b1) {
  long i = (long)blockIdx.x * blockDim.x + threadIdx.x;
  long st = (long)gridDim.x * blockDim.x;
  for (long k = i; k < (long)G4H * EMB; k += st) wih0[k] = (bf16t)Wih0[k];
  for (long k = i; k < (long)G4H * HID; k += st) whh0[k] = (bf16t)Whh0[k];
  for (long k = i; k < (long)G4H * HID; k += st) wih1[k] = (bf16t)Wih1[k];
  for (long k = i; k < (long)G4H * HID; k += st) whh1[k] = (bf16t)Whh1[k];
  for (long k = i; k < (long)VOC * HID; k += st) wfc[k] = (bf16t)Wfc[k];
  for (long k = i; k < G4H; k += st) {
    b0[k] = bih0[k] + bhh0[k];
    b1[k] = bih1[k] + bhh1[k];
  }
}

// ---------- layer-0 input projection: xp0[t*16+b, :] = emb[x[b,t]] @ Wih0^T + b0 ----------
__global__ __launch_bounds__(256) void k_xproj0(const int* x, const float* emb,
                                                const bf16t* wih0, const float* b0,
                                                float* xp0) {
  int lane = threadIdx.x & 31;
  int wave = blockIdx.x * (blockDim.x >> 5) + (threadIdx.x >> 5);
  const int NCH = G4H / 64;  // 16 chunks of 64 cols
  int t = wave / NCH;
  if (t >= SEQ) return;
  int n0 = (wave % NCH) * 64;
  int bb = lane & 15;
  int tok = x[bb * SEQ + t];
  const float* arow = emb + (long)tok * EMB;

  v8f acc[4];
#pragma unroll
  for (int j = 0; j < 4; ++j)
#pragma unroll
    for (int r = 0; r < 8; ++r) acc[j][r] = 0.f;

  for (int k0 = 0; k0 < EMB; k0 += 32) {
    v16bf a = fragA_f32(arow, k0, lane);
#pragma unroll
    for (int j = 0; j < 4; ++j) {
      v16bf bm = fragB(wih0, n0 + j * 16 + (lane & 15), EMB, k0, lane);
      acc[j] = wmma_bf16(a, bm, acc[j]);
    }
  }
  int mo = (lane & 16) ? 8 : 0;
  int nc = lane & 15;
#pragma unroll
  for (int j = 0; j < 4; ++j) {
    int n = n0 + j * 16 + nc;
    float bias = b0[n];
#pragma unroll
    for (int r = 0; r < 8; ++r) {
      int brow = r + mo;
      xp0[((long)t * BAT + brow) * G4H + n] = acc[j][r] + bias;
    }
  }
}

// ---------- layer-1 input projection: xp1 = h1s @ Wih1^T + b1 (A already bf16) ----------
__global__ __launch_bounds__(256) void k_xproj1(const bf16t* h1s, const bf16t* wih1,
                                                const float* b1, float* xp1) {
  int lane = threadIdx.x & 31;
  int wave = blockIdx.x * (blockDim.x >> 5) + (threadIdx.x >> 5);
  const int NCH = G4H / 64;
  int mt = wave / NCH;
  if (mt >= (SEQ * BAT) / 16) return;
  int n0 = (wave % NCH) * 64;
  int row0 = mt * 16;

  v8f acc[4];
#pragma unroll
  for (int j = 0; j < 4; ++j)
#pragma unroll
    for (int r = 0; r < 8; ++r) acc[j][r] = 0.f;

  for (int k0 = 0; k0 < HID; k0 += 32) {
    v16bf a = fragA(h1s, row0 + (lane & 15), HID, k0, lane);
#pragma unroll
    for (int j = 0; j < 4; ++j) {
      v16bf bm = fragB(wih1, n0 + j * 16 + (lane & 15), HID, k0, lane);
      acc[j] = wmma_bf16(a, bm, acc[j]);
    }
  }
  int mo = (lane & 16) ? 8 : 0;
  int nc = lane & 15;
#pragma unroll
  for (int j = 0; j < 4; ++j) {
    int n = n0 + j * 16 + nc;
    float bias = b1[n];
#pragma unroll
    for (int r = 0; r < 8; ++r) {
      xp1[((long)row0 + r + mo) * G4H + n] = acc[j][r] + bias;
    }
  }
}

// ---------- recurrent LSTM layer: single workgroup, 16 waves ----------
// wave w owns hidden slice n in [w*16, w*16+16); computes all 4 gate tiles so the
// cell state c stays in accumulator-layout registers across all T steps.
__global__ __launch_bounds__(512) void k_lstm(const float* xp,     // [T*B, 4H]
                                              const bf16t* whh,    // [4H, H]
                                              bf16t* hs,           // [T*B, H]
                                              float* hT, float* cT) {
  __shared__ __align__(32) bf16t hbuf[BAT * HID];
  int lane = threadIdx.x & 31;
  int wave = threadIdx.x >> 5;       // 0..15
  int nloc = wave * 16 + (lane & 15);
  int mo = (lane & 16) ? 8 : 0;

  for (int i = threadIdx.x; i < BAT * HID; i += blockDim.x) hbuf[i] = (bf16t)0.f;
  float creg[8];
#pragma unroll
  for (int r = 0; r < 8; ++r) creg[r] = 0.f;
  __syncthreads();

  for (int t = 0; t < SEQ; ++t) {
    v8f acc[4];
#pragma unroll
    for (int g = 0; g < 4; ++g)
#pragma unroll
      for (int r = 0; r < 8; ++r)
        acc[g][r] = xp[((long)t * BAT + (r + mo)) * G4H + g * HID + nloc];

    for (int k0 = 0; k0 < HID; k0 += 32) {
      v16bf a = fragA(hbuf, lane & 15, HID, k0, lane);
#pragma unroll
      for (int g = 0; g < 4; ++g) {
        v16bf bm = fragB(whh, g * HID + nloc, HID, k0, lane);
        acc[g] = wmma_bf16(a, bm, acc[g]);
      }
    }
    __syncthreads();  // all waves done reading h(t-1)
#pragma unroll
    for (int r = 0; r < 8; ++r) {
      float iv = sigm(acc[0][r]);
      float fv = sigm(acc[1][r]);
      float gv = fast_tanh(acc[2][r]);
      float ov = sigm(acc[3][r]);
      float cc = fv * creg[r] + iv * gv;
      creg[r] = cc;
      float hv = ov * fast_tanh(cc);
      int brow = r + mo;
      hbuf[brow * HID + nloc] = (bf16t)hv;
      hs[((long)t * BAT + brow) * HID + nloc] = (bf16t)hv;
      if (t == SEQ - 1) {
        hT[brow * HID + nloc] = hv;
        cT[brow * HID + nloc] = cc;
      }
    }
    __syncthreads();  // h(t) visible for next step
  }
}

// ---------- vocab head: logits[b,t,:] = h2s[t*16+b,:] @ Wfc^T + b_fc ----------
// Macro-tile 128(M) x 64(N) per workgroup: the 64x256 bf16 W_fc tile (32 KB) is
// staged in LDS once via CDNA5 async global->LDS copies, then 8 waves each
// compute one 16-row tile reading B from LDS (8x less L2 traffic on W_fc).
__global__ __launch_bounds__(256) void k_logits(const bf16t* h2s, const bf16t* wfc,
                                                const float* bfc, float* out) {
  __shared__ __align__(32) bf16t btile[64 * HID];  // 32 KB
  int lane = threadIdx.x & 31;
  int wave = threadIdx.x >> 5;  // 0..7
  const int NCH = VOC / 64;     // 500
  int mg = blockIdx.x / NCH;    // 0..31  (M group of 128 rows)
  int nc = blockIdx.x % NCH;
  int n0 = nc * 64;
  int row0 = mg * 128 + wave * 16;

  // --- async fill of btile: 64 rows x 256 cols bf16 = 2048 x 16B chunks ---
  {
    unsigned lbase = (unsigned)(unsigned long long)(&btile[0]);
    unsigned long long gbase =
        (unsigned long long)(wfc + (unsigned long long)n0 * HID);
#pragma unroll
    for (int i = 0; i < 8; ++i) {
      unsigned c = threadIdx.x + i * 256;          // chunk id, 16 bytes each
      unsigned loff = lbase + c * 16;
      unsigned long long ga = gbase + (unsigned long long)c * 16;
      asm volatile("global_load_async_to_lds_b128 %0, %1, off"
                   :
                   : "v"(loff), "v"(ga)
                   : "memory");
    }
    asm volatile("s_wait_asynccnt 0x0" ::: "memory");
  }
  __syncthreads();

  v8f acc[4];
#pragma unroll
  for (int j = 0; j < 4; ++j)
#pragma unroll
    for (int r = 0; r < 8; ++r) acc[j][r] = 0.f;

  for (int k0 = 0; k0 < HID; k0 += 32) {
    v16bf a = fragA(h2s, row0 + (lane & 15), HID, k0, lane);
#pragma unroll
    for (int j = 0; j < 4; ++j) {
      v16bf bm = fragB(btile, j * 16 + (lane & 15), HID, k0, lane);
      acc[j] = wmma_bf16(a, bm, acc[j]);
    }
  }
  int mo = (lane & 16) ? 8 : 0;
  int ncl = lane & 15;
  int t = row0 >> 4;  // tile row == timestep (rows are t*16+b)
#pragma unroll
  for (int j = 0; j < 4; ++j) {
    int n = n0 + j * 16 + ncl;
    float bias = bfc[n];
#pragma unroll
    for (int r = 0; r < 8; ++r) {
      int bb = r + mo;
      out[((long)bb * SEQ + t) * VOC + n] = acc[j][r] + bias;
    }
  }
}

extern "C" void kernel_launch(void* const* d_in, const int* in_sizes, int n_in,
                              void* d_out, int out_size, void* d_ws, size_t ws_size,
                              hipStream_t stream) {
  (void)in_sizes; (void)n_in; (void)out_size; (void)ws_size;
  const int*   x    = (const int*)d_in[0];
  const float* emb  = (const float*)d_in[1];
  const float* Wih0 = (const float*)d_in[2];
  const float* Whh0 = (const float*)d_in[3];
  const float* bih0 = (const float*)d_in[4];
  const float* bhh0 = (const float*)d_in[5];
  const float* Wih1 = (const float*)d_in[6];
  const float* Whh1 = (const float*)d_in[7];
  const float* bih1 = (const float*)d_in[8];
  const float* bhh1 = (const float*)d_in[9];
  const float* Wfc  = (const float*)d_in[10];
  const float* bfc  = (const float*)d_in[11];

  char* w = (char*)d_ws;
  size_t off = 0;
  auto carve = [&](size_t bytes) -> void* {
    void* p = w + off;
    off += (bytes + 255) & ~(size_t)255;
    return p;
  };
  bf16t* wih0 = (bf16t*)carve((size_t)G4H * EMB * 2);
  bf16t* whh0 = (bf16t*)carve((size_t)G4H * HID * 2);
  bf16t* wih1 = (bf16t*)carve((size_t)G4H * HID * 2);
  bf16t* whh1 = (bf16t*)carve((size_t)G4H * HID * 2);
  bf16t* wfc  = (bf16t*)carve((size_t)VOC * HID * 2);
  float* b0   = (float*)carve((size_t)G4H * 4);
  float* b1   = (float*)carve((size_t)G4H * 4);
  float* xp0  = (float*)carve((size_t)SEQ * BAT * G4H * 4);
  float* xp1  = (float*)carve((size_t)SEQ * BAT * G4H * 4);
  bf16t* h1s  = (bf16t*)carve((size_t)SEQ * BAT * HID * 2);
  bf16t* h2s  = (bf16t*)carve((size_t)SEQ * BAT * HID * 2);

  float* outp   = (float*)d_out;
  float* logits = outp;
  float* hn     = outp + (size_t)BAT * SEQ * VOC;  // [2,B,H]
  float* cn     = hn + 2 * BAT * HID;              // [2,B,H]

  k_prep<<<dim3(2048), dim3(256), 0, stream>>>(Wih0, Whh0, Wih1, Whh1, Wfc, bih0,
                                               bhh0, bih1, bhh1, wih0, whh0, wih1,
                                               whh1, wfc, b0, b1);

  // 4096 wave-tiles, 8 waves per 256-thread block -> 512 blocks
  k_xproj0<<<dim3(512), dim3(256), 0, stream>>>(x, emb, wih0, b0, xp0);
  k_lstm<<<dim3(1), dim3(512), 0, stream>>>(xp0, whh0, h1s, hn, cn);
  k_xproj1<<<dim3(512), dim3(256), 0, stream>>>(h1s, wih1, b1, xp1);
  k_lstm<<<dim3(1), dim3(512), 0, stream>>>(xp1, whh1, h2s, hn + BAT * HID,
                                            cn + BAT * HID);
  // 32 M-groups (128 rows) * 500 N-chunks -> 16000 blocks of 8 waves
  k_logits<<<dim3(16000), dim3(256), 0, stream>>>(h2s, wfc, bfc, logits);
}
